// StateSpaceModel_88381837017816
// MI455X (gfx1250) — compile-verified, matrix operand order checked by reading, not run on hardware
//
#include <hip/hip_runtime.h>
#include <math.h>

#define BATCH 32
#define SEQ   2048
#define IDIM  512
#define SDIM  256
#define ODIM  256
#define NCHUNK 32
#define CLEN   64   // SEQ / NCHUNK

#define SG_BYTES     (SDIM * SDIM * 2)          // 128 KB: G staged in LDS
#define STATE_BYTES  (2 * BATCH * SDIM * 2)     // 32 KB: double-buffered state
#define LDS_BYTES    (SG_BYTES + STATE_BYTES)   // 160 KB dynamic LDS

typedef __bf16 bf16_t;
typedef __attribute__((ext_vector_type(16))) __bf16 v16bf;
typedef __attribute__((ext_vector_type(8)))  float  v8f;
typedef __attribute__((ext_vector_type(4)))  unsigned int u32x4;
typedef __attribute__((ext_vector_type(8)))  int i32x8;
typedef __attribute__((ext_vector_type(4)))  int i32x4;

__device__ __forceinline__ v8f wmma_bf16(v16bf a, v16bf b, v8f c) {
  return __builtin_amdgcn_wmma_f32_16x16x32_bf16(false, a, false, b, (short)0, c,
                                                 false, false);
}

__device__ __forceinline__ v16bf ldcvt16(const float* __restrict__ p) {
  v16bf r;
#pragma unroll
  for (int i = 0; i < 16; ++i) r[i] = (__bf16)p[i];
  return r;
}

// TDM: async-DMA a 256x256 bf16 tensor from global into LDS (one call per wave).
// D# per CDNA5 ISA. 6-arg builtin variant.
__device__ __forceinline__ void tdm_load_256x256_bf16(const bf16_t* gsrc, bf16_t* ldst) {
  unsigned long long ga = (unsigned long long)(uintptr_t)gsrc;
  unsigned int la = (unsigned int)(uintptr_t)ldst;   // LDS aperture: low 32 bits = LDS byte addr
  u32x4 g0;
  g0[0] = 1u;                                         // count=1 (valid descriptor)
  g0[1] = la;                                         // lds_addr (bytes)
  g0[2] = (unsigned int)ga;                           // global_addr[31:0]
  g0[3] = (unsigned int)((ga >> 32) & 0x1FFFFFFu) | (2u << 30);  // addr[56:32] | type=2
  i32x8 g1;
  g1[0] = 0x10000;                                    // data_size=1 -> 2-byte elements
  g1[1] = (int)(256u << 16);                          // tensor_dim0 = 256 (low 16 bits)
  g1[2] = (int)(256u << 16);                          // tensor_dim1 = 256 (low 16 bits)
  g1[3] = (int)(256u << 16);                          // tile_dim0 = 256
  g1[4] = 256;                                        // tile_dim1 = 256 (tile_dim2 = 0)
  g1[5] = 256;                                        // tensor_dim0_stride = 256 (low 32)
  g1[6] = 0;
  g1[7] = 0;
  i32x4 z4 = {};
  i32x8 z8 = {};
  __builtin_amdgcn_tensor_load_to_lds(g0, g1, z4, z4, z8, 0);
}

// x_t * B^T contribution (K = IDIM), accumulated into acc[0..3]
__device__ __forceinline__ void accum_bx(const float* __restrict__ xt,
                                         const bf16_t* __restrict__ Bs,
                                         int ng, int lo, int hi, v8f acc[4]) {
#pragma unroll
  for (int kk = 0; kk < IDIM / 32; ++kk) {
    v16bf a = ldcvt16(xt + kk * 32 + hi * 16);
#pragma unroll
    for (int j = 0; j < 4; ++j) {
      int n = ng * 4 + j;
      v16bf b = *(const v16bf*)(Bs + (n * 16 + lo) * IDIM + kk * 32 + hi * 16);
      acc[j] = wmma_bf16(a, b, acc[j]);
    }
  }
}

// ---------------------------------------------------------------------------
// prep: G = tanh(A) (row + transposed col copies), bf16 copies of B, C, D
// ---------------------------------------------------------------------------
__global__ void ssm_prep_kernel(const float* __restrict__ A, const float* __restrict__ B,
                                const float* __restrict__ C, const float* __restrict__ D,
                                bf16_t* __restrict__ Bs, bf16_t* __restrict__ Ds,
                                bf16_t* __restrict__ Cb, bf16_t* __restrict__ Grow,
                                bf16_t* __restrict__ Gcol) {
  int id = blockIdx.x * blockDim.x + threadIdx.x;   // covers 256*512
  if (id < SDIM * IDIM) {
    Bs[id] = (bf16_t)B[id];
    Ds[id] = (bf16_t)D[id];
  }
  if (id < SDIM * SDIM) {
    float a = tanhf(A[id]);
    Grow[id] = (bf16_t)a;
    int r = id >> 8, c = id & 255;
    Gcol[c * SDIM + r] = (bf16_t)a;
    Cb[id] = (bf16_t)C[id];
  }
}

// ---------------------------------------------------------------------------
// square: H = G*G (256x256), writes row-major and transposed copies
// grid 64 x block 128 (4 waves, 1 output tile per wave)
// ---------------------------------------------------------------------------
__global__ void ssm_square_kernel(const bf16_t* __restrict__ Gin_row,
                                  const bf16_t* __restrict__ Gin_col,
                                  bf16_t* __restrict__ Gout_row,
                                  bf16_t* __restrict__ Gout_col) {
  const int lane = threadIdx.x & 31;
  const int w    = threadIdx.x >> 5;
  const int tile = blockIdx.x * 4 + w;     // 0..255
  const int tm = tile >> 4, tn = tile & 15;
  const int lo = lane & 15, hi = lane >> 4;
  v8f acc = {};
#pragma unroll
  for (int kk = 0; kk < SDIM / 32; ++kk) {
    v16bf a = *(const v16bf*)(Gin_row + (tm * 16 + lo) * SDIM + kk * 32 + hi * 16);
    v16bf b = *(const v16bf*)(Gin_col + (tn * 16 + lo) * SDIM + kk * 32 + hi * 16);
    acc = wmma_bf16(a, b, acc);
  }
#pragma unroll
  for (int r = 0; r < 8; ++r) {
    int row = tm * 16 + r + 8 * hi;
    int col = tn * 16 + lo;
    bf16_t v = (bf16_t)acc[r];
    Gout_row[row * SDIM + col] = v;
    Gout_col[col * SDIM + row] = v;
  }
}

// ---------------------------------------------------------------------------
// chunk scan: block c owns timesteps [c*CLEN, (c+1)*CLEN).
// G staged in LDS via TDM; state double-buffered in LDS (1 barrier/step);
// next step's x*B^T issued into the barrier shadow.
// grid NCHUNK x block 256 (8 waves)
// ---------------------------------------------------------------------------
__global__ void ssm_chunk_scan_kernel(const float* __restrict__ x,
                                      const bf16_t* __restrict__ Bs,
                                      const bf16_t* __restrict__ G,
                                      float* __restrict__ partials) {
  extern __shared__ char smem[];
  bf16_t* s_G     = (bf16_t*)smem;                       // 128 KB
  bf16_t* s_state = (bf16_t*)(smem + SG_BYTES);          // 2 x 16 KB

  const int c    = blockIdx.x;
  const int lane = threadIdx.x & 31;
  const int w    = threadIdx.x >> 5;   // 0..7
  const int mt   = w >> 2;             // batch tile 0..1
  const int ng   = w & 3;              // n-tile group 0..3
  const int lo = lane & 15, hi = lane >> 4;

  if (w == 0) tdm_load_256x256_bf16(G, s_G);             // async DMA, TENSORcnt

  // zero state buffer 0 (the t=0 read buffer)
  for (int i = threadIdx.x; i < BATCH * SDIM; i += blockDim.x)
    s_state[i] = (bf16_t)0.f;

  const int b0 = mt * 16 + lo;
  const float* xrow = x + ((size_t)b0 * SEQ + (size_t)c * CLEN) * IDIM;

  // bx(0) while the TDM transfer is in flight
  v8f acc[4] = {};
  accum_bx(xrow, Bs, ng, lo, hi, acc);

  if (w == 0) __builtin_amdgcn_s_wait_tensorcnt(0);
  __syncthreads();                                        // s_G + zeroed state visible

  for (int t = 0; t < CLEN; ++t) {
    const bf16_t* srd = s_state + (t & 1) * (BATCH * SDIM);
    bf16_t*       swr = s_state + ((t + 1) & 1) * (BATCH * SDIM);

    // recurrence: acc += state * At^T  (K = SDIM, operands from LDS)
#pragma unroll
    for (int kk = 0; kk < SDIM / 32; ++kk) {
      v16bf a = *(const v16bf*)(srd + (mt * 16 + lo) * SDIM + kk * 32 + hi * 16);
#pragma unroll
      for (int j = 0; j < 4; ++j) {
        int n = ng * 4 + j;
        v16bf b = *(const v16bf*)(s_G + (n * 16 + lo) * SDIM + kk * 32 + hi * 16);
        acc[j] = wmma_bf16(a, b, acc[j]);
      }
    }

    // issue next step's independent x*B^T into the barrier shadow
    v8f nxt[4] = {};
    if (t + 1 < CLEN) {
      const float* xt1 = xrow + (size_t)(t + 1) * IDIM;
      if (t + 2 < CLEN) __builtin_prefetch(xt1 + IDIM, 0, 0);  // global_prefetch_b8
      accum_bx(xt1, Bs, ng, lo, hi, nxt);
    }

    // publish new state (write buffer != read buffer -> single barrier)
#pragma unroll
    for (int j = 0; j < 4; ++j) {
      int n = ng * 4 + j;
#pragma unroll
      for (int r = 0; r < 8; ++r)
        swr[(mt * 16 + r + 8 * hi) * SDIM + n * 16 + lo] = (bf16_t)acc[j][r];
    }
    if (t == CLEN - 1) {
      float* P = partials + (size_t)c * (BATCH * SDIM);
#pragma unroll
      for (int j = 0; j < 4; ++j) {
        int n = ng * 4 + j;
#pragma unroll
        for (int r = 0; r < 8; ++r)
          P[(mt * 16 + r + 8 * hi) * SDIM + n * 16 + lo] = acc[j][r];
      }
    }
    __syncthreads();
#pragma unroll
    for (int j = 0; j < 4; ++j) acc[j] = nxt[j];
  }
}

// ---------------------------------------------------------------------------
// combine: F = ((P0*Mp + P1)*Mp + P2)... then out = F*C^T + x_last*D^T
// grid 1 x block 256; Mp^T staged in LDS via TDM; state double-buffered.
// ---------------------------------------------------------------------------
__global__ void ssm_combine_kernel(const float* __restrict__ x,
                                   const float* __restrict__ partials,
                                   const bf16_t* __restrict__ Gp,
                                   const bf16_t* __restrict__ Cb,
                                   const bf16_t* __restrict__ Ds,
                                   float* __restrict__ out) {
  extern __shared__ char smem[];
  bf16_t* s_G     = (bf16_t*)smem;
  bf16_t* s_state = (bf16_t*)(smem + SG_BYTES);

  const int lane = threadIdx.x & 31;
  const int w    = threadIdx.x >> 5;
  const int mt = w >> 2, ng = w & 3;
  const int lo = lane & 15, hi = lane >> 4;

  if (w == 0) tdm_load_256x256_bf16(Gp, s_G);

  for (int i = threadIdx.x; i < BATCH * SDIM; i += blockDim.x)
    s_state[i] = (bf16_t)partials[i];                    // F = P_0 into buffer 0

  if (w == 0) __builtin_amdgcn_s_wait_tensorcnt(0);
  __syncthreads();

  for (int cc = 1; cc < NCHUNK; ++cc) {
    const bf16_t* srd = s_state + ((cc - 1) & 1) * (BATCH * SDIM);
    bf16_t*       swr = s_state + (cc & 1) * (BATCH * SDIM);
    const float* P = partials + (size_t)cc * (BATCH * SDIM);
    v8f acc[4];
#pragma unroll
    for (int j = 0; j < 4; ++j) {
      int n = ng * 4 + j;
#pragma unroll
      for (int r = 0; r < 8; ++r)
        acc[j][r] = P[(mt * 16 + r + 8 * hi) * SDIM + n * 16 + lo];
    }
#pragma unroll
    for (int kk = 0; kk < SDIM / 32; ++kk) {
      v16bf a = *(const v16bf*)(srd + (mt * 16 + lo) * SDIM + kk * 32 + hi * 16);
#pragma unroll
      for (int j = 0; j < 4; ++j) {
        int n = ng * 4 + j;
        v16bf b = *(const v16bf*)(s_G + (n * 16 + lo) * SDIM + kk * 32 + hi * 16);
        acc[j] = wmma_bf16(a, b, acc[j]);
      }
    }
#pragma unroll
    for (int j = 0; j < 4; ++j) {
      int n = ng * 4 + j;
#pragma unroll
      for (int r = 0; r < 8; ++r)
        swr[(mt * 16 + r + 8 * hi) * SDIM + n * 16 + lo] = (bf16_t)acc[j][r];
    }
    __syncthreads();
  }

  // out = F*C^T + x_last*D^T  (F is in buffer (NCHUNK-1)&1)
  const bf16_t* F = s_state + ((NCHUNK - 1) & 1) * (BATCH * SDIM);
  const int b0 = mt * 16 + lo;
  const float* xlast = x + ((size_t)b0 * SEQ + (SEQ - 1)) * IDIM;
  v8f acc[4] = {};
#pragma unroll
  for (int kk = 0; kk < SDIM / 32; ++kk) {
    v16bf a = *(const v16bf*)(F + (mt * 16 + lo) * SDIM + kk * 32 + hi * 16);
#pragma unroll
    for (int j = 0; j < 4; ++j) {
      int n = ng * 4 + j;
      v16bf b = *(const v16bf*)(Cb + (n * 16 + lo) * SDIM + kk * 32 + hi * 16);
      acc[j] = wmma_bf16(a, b, acc[j]);
    }
  }
  accum_bx(xlast, Ds, ng, lo, hi, acc);
#pragma unroll
  for (int j = 0; j < 4; ++j) {
    int n = ng * 4 + j;
#pragma unroll
    for (int r = 0; r < 8; ++r)
      out[(mt * 16 + r + 8 * hi) * ODIM + n * 16 + lo] = acc[j][r];
  }
}

// ---------------------------------------------------------------------------
extern "C" void kernel_launch(void* const* d_in, const int* in_sizes, int n_in,
                              void* d_out, int out_size, void* d_ws, size_t ws_size,
                              hipStream_t stream) {
  (void)in_sizes; (void)n_in; (void)out_size; (void)ws_size;
  const float* x = (const float*)d_in[0];
  const float* A = (const float*)d_in[1];
  const float* B = (const float*)d_in[2];
  const float* C = (const float*)d_in[3];
  const float* D = (const float*)d_in[4];
  float* out = (float*)d_out;

  char* ws = (char*)d_ws;                       // ~2.4 MB used
  bf16_t* Bs   = (bf16_t*)(ws + 0);             // 256*512*2
  bf16_t* Dsb  = (bf16_t*)(ws + 262144);        // 256*512*2
  bf16_t* Cbf  = (bf16_t*)(ws + 524288);        // 256*256*2
  bf16_t* Grow = (bf16_t*)(ws + 655360);        // 256*256*2
  bf16_t* Gcol = (bf16_t*)(ws + 786432);
  bf16_t* Prow = (bf16_t*)(ws + 917504);
  bf16_t* Pcol = (bf16_t*)(ws + 1048576);
  bf16_t* Qrow = (bf16_t*)(ws + 1179648);
  bf16_t* Qcol = (bf16_t*)(ws + 1310720);
  float*  Part = (float*)(ws + 1441792);        // 32*32*256*4

  ssm_prep_kernel<<<512, 256, 0, stream>>>(A, B, C, D, Bs, Dsb, Cbf, Grow, Gcol);

  // 6 squarings: M -> M^64 (in transposed G-representation)
  const bf16_t* inr = Grow; const bf16_t* inc = Gcol;
  bf16_t* our = Prow; bf16_t* ouc = Pcol;
  bf16_t* alr = Qrow; bf16_t* alc = Qcol;
  for (int i = 0; i < 6; ++i) {
    ssm_square_kernel<<<64, 128, 0, stream>>>(inr, inc, our, ouc);
    inr = our; inc = ouc;
    bf16_t* tr = our; our = alr; alr = tr;
    bf16_t* tc = ouc; ouc = alc; alc = tc;
  }
  const bf16_t* Gp = inr;   // (M^64)^T, row-major

  ssm_chunk_scan_kernel<<<NCHUNK, 256, LDS_BYTES, stream>>>(x, Bs, Grow, Part);
  ssm_combine_kernel<<<1, 256, LDS_BYTES, stream>>>(x, Part, Gp, Cbf, Dsb, out);
}
